// cross_bn_PointTransformerLayer_25220047962575
// MI455X (gfx1250) — matrix operand order repaired and checked by reading, not compiled, not run on hardware
//
#include <hip/hip_runtime.h>
#include <hip/hip_bf16.h>

typedef _Float16 half16 __attribute__((ext_vector_type(16)));
typedef float    float8 __attribute__((ext_vector_type(8)));

#define Bn    4
#define Nn    4096
#define Kn    16
#define INn   64
#define OUTn  128
#define MIDn  64
#define NPTS  (Bn * Nn)          // 16384
#define EPSf  1e-5f

// ---------------------------------------------------------------------------
// Kernel 1: fused q/k/v projections via V_WMMA_F32_16X16X32_F16.
// grid = (NPTS/16, 16): blockIdx.y 0-3 -> q tiles, 4-7 -> k tiles, 8-15 -> v tiles.
// One wave (32 lanes) produces one 16x16 f32 output tile; Kdim=64 = two k-steps.
// ---------------------------------------------------------------------------
__global__ __launch_bounds__(32)
void proj_wmma_kernel(const float* __restrict__ points1,
                      const float* __restrict__ points2,
                      const float* __restrict__ wq, const float* __restrict__ bq,
                      const float* __restrict__ wk, const float* __restrict__ bk,
                      const float* __restrict__ wv, const float* __restrict__ bv,
                      float* __restrict__ xq, float* __restrict__ xk,
                      float* __restrict__ xv)
{
    const int lane    = threadIdx.x;       // 0..31
    const int rowTile = blockIdx.x;        // 0..1023
    const int ty      = blockIdx.y;        // 0..15

    const float* src; const float* W; const float* bias; float* dst;
    int ncols, colBase;
    if (ty < 4)      { src = points1; W = wq; bias = bq; dst = xq; ncols = MIDn; colBase = ty * 16; }
    else if (ty < 8) { src = points2; W = wk; bias = bk; dst = xk; ncols = MIDn; colBase = (ty - 4) * 16; }
    else             { src = points2; W = wv; bias = bv; dst = xv; ncols = OUTn; colBase = (ty - 8) * 16; }

    const int r16  = lane & 15;
    const int half = lane >> 4;                    // which 16-lane group
    const int arow = rowTile * 16 + r16;           // A row held by this lane
    const int col  = colBase + r16;                // B/C column held by this lane

    float8 acc = {};
    #pragma unroll
    for (int kk = 0; kk < 2; ++kk) {
        // A fragment: 16x32 f16. lane-group h holds
        //   elements e<8 : K = kk*32 + h*16 + e
        //   elements e>=8: K = kk*32 + h*16 + e
        half16 a;
        const float* ap = src + arow * INn + kk * 32 + half * 16;
        #pragma unroll
        for (int e = 0; e < 8; ++e)  a[e]     = (_Float16)ap[e];
        #pragma unroll
        for (int e = 0; e < 8; ++e)  a[e + 8] = (_Float16)ap[8 + e];

        // B fragment: 32x16 f16. lane-group h holds K = kk*32 + h*16 + e at column (lane&15).
        half16 b;
        const float* bp = W + (kk * 32 + half * 16) * ncols + col;
        #pragma unroll
        for (int e = 0; e < 16; ++e) b[e] = (_Float16)bp[e * ncols];

        acc = __builtin_amdgcn_wmma_f32_16x16x32_f16(
                  false, a, false, b, (short)0, acc, false, false);
    }

    const float bc = bias[col];
    #pragma unroll
    for (int r = 0; r < 8; ++r) {
        const int orow = rowTile * 16 + r + half * 8;   // C layout: vgpr r, lane-group h
        dst[orow * ncols + col] = acc[r] + bc;
    }
}

// ---------------------------------------------------------------------------
// Kernel 2: brute-force KNN (top-16 by squared distance), one thread per query.
// xyz2 streamed through LDS in 256-point tiles; register-resident replace-max
// top-16; final unrolled sort ascending by (dist, idx) to match top_k order.
// ---------------------------------------------------------------------------
#define TS 256
__global__ __launch_bounds__(256)
void knn_kernel(const float* __restrict__ xyz1,
                const float* __restrict__ xyz2,
                int* __restrict__ idxbuf)
{
    __shared__ float sx[TS], sy[TS], sz[TS];
    const int tid   = threadIdx.x;
    const int batch = blockIdx.x >> 4;               // 16 blocks per batch
    const int qi    = ((blockIdx.x & 15) << 8) + tid;
    const int q     = batch * Nn + qi;

    const float qx = xyz1[q * 3 + 0];
    const float qy = xyz1[q * 3 + 1];
    const float qz = xyz1[q * 3 + 2];

    float bd[Kn]; int bi[Kn];
    #pragma unroll
    for (int i = 0; i < Kn; ++i) { bd[i] = 3.4e38f; bi[i] = 0; }
    float wmax = 3.4e38f;

    const float* cand = xyz2 + (size_t)batch * Nn * 3;
    for (int t0 = 0; t0 < Nn; t0 += TS) {
        __syncthreads();
        sx[tid] = cand[(t0 + tid) * 3 + 0];
        sy[tid] = cand[(t0 + tid) * 3 + 1];
        sz[tid] = cand[(t0 + tid) * 3 + 2];
        __syncthreads();
        if (t0 + TS < Nn)
            __builtin_prefetch(&cand[(t0 + TS + tid) * 3], 0, 0);

        for (int c = 0; c < TS; ++c) {
            const float dx = qx - sx[c];
            const float dy = qy - sy[c];
            const float dz = qz - sz[c];
            const float d  = dx * dx + dy * dy + dz * dz;
            if (d < wmax) {                       // amortized-rare path
                bool done = false;
                #pragma unroll
                for (int i = 0; i < Kn; ++i) {
                    if (!done && bd[i] == wmax) { bd[i] = d; bi[i] = t0 + c; done = true; }
                }
                wmax = bd[0];
                #pragma unroll
                for (int i = 1; i < Kn; ++i) wmax = fmaxf(wmax, bd[i]);
            }
        }
    }

    // fully-unrolled stable sort: ascending dist, ties -> lower index first
    #pragma unroll
    for (int i = 0; i < Kn - 1; ++i) {
        #pragma unroll
        for (int j = 0; j < Kn - 1; ++j) {
            const float d0 = bd[j], d1 = bd[j + 1];
            const int   i0 = bi[j], i1 = bi[j + 1];
            const bool sw = (d0 > d1) || (d0 == d1 && i0 > i1);
            bd[j] = sw ? d1 : d0;  bd[j + 1] = sw ? d0 : d1;
            bi[j] = sw ? i1 : i0;  bi[j + 1] = sw ? i0 : i1;
        }
    }

    #pragma unroll
    for (int k = 0; k < Kn; ++k) idxbuf[q * Kn + k] = bi[k];
}

// ---------------------------------------------------------------------------
// Kernel 3: fused gather + linear_p + logits + linear_w + softmax + weighting.
// 256 threads = 16 points x 16 neighbors. Softmax over K via shfl_xor(width=16)
// inside wave32 half-waves. All weights / BN affines staged in LDS.
// ---------------------------------------------------------------------------
__global__ __launch_bounds__(256)
void attn_kernel(const float* __restrict__ xyz1, const float* __restrict__ xyz2,
                 const float* __restrict__ xq, const float* __restrict__ xk,
                 const float* __restrict__ xv, const int* __restrict__ idxbuf,
                 const float* __restrict__ wp1, const float* __restrict__ bp1,
                 const float* __restrict__ pg,  const float* __restrict__ pb,
                 const float* __restrict__ pm,  const float* __restrict__ pv,
                 const float* __restrict__ wp2, const float* __restrict__ bp2,
                 const float* __restrict__ b1g, const float* __restrict__ b1b,
                 const float* __restrict__ b1m, const float* __restrict__ b1v,
                 const float* __restrict__ ww1, const float* __restrict__ bw1,
                 const float* __restrict__ b2g, const float* __restrict__ b2b,
                 const float* __restrict__ b2m, const float* __restrict__ b2v,
                 const float* __restrict__ ww2, const float* __restrict__ bw2,
                 float* __restrict__ outx, float* __restrict__ outp)
{
    __shared__ float s_wp2[3 * OUTn], s_bp2[OUTn];
    __shared__ float s_ww1[MIDn * 8], s_ww2[8 * 16];
    __shared__ float s_s1[MIDn], s_t1[MIDn];
    __shared__ float s_bw1[8], s_s2[8], s_t2[8], s_bw2[16];
    __shared__ float s_xq[16][MIDn];

    const int tid    = threadIdx.x;
    const int ptBase = blockIdx.x * 16;

    for (int i = tid; i < 3 * OUTn; i += 256) s_wp2[i] = wp2[i];
    for (int i = tid; i < MIDn * 8; i += 256) s_ww1[i] = ww1[i];
    if (tid < OUTn) s_bp2[tid] = bp2[tid];
    if (tid < 128)  { /* ww2 is 8*16 = 128 */ s_ww2[tid & 127] = ww2[tid & 127]; }
    if (tid < MIDn) {
        const float s = b1g[tid] * rsqrtf(b1v[tid] + EPSf);
        s_s1[tid] = s;  s_t1[tid] = b1b[tid] - b1m[tid] * s;
    }
    if (tid < 8) {
        const float s = b2g[tid] * rsqrtf(b2v[tid] + EPSf);
        s_s2[tid] = s;  s_t2[tid] = b2b[tid] - b2m[tid] * s;
        s_bw1[tid] = bw1[tid];
    }
    if (tid < 16) s_bw2[tid] = bw2[tid];
    for (int i = tid; i < 16 * MIDn; i += 256) {
        const int lp = i >> 6, c = i & 63;
        s_xq[lp][c] = xq[(ptBase + lp) * MIDn + c];
    }
    __syncthreads();

    const int lp    = tid >> 4;               // local point 0..15
    const int kk    = tid & 15;               // neighbor 0..15
    const int point = ptBase + lp;
    const int batch = point >> 12;            // /4096
    const int nb    = idxbuf[point * Kn + kk];
    const int g     = batch * Nn + nb;

    // relative position p_r0 (second tuple output)
    const float prx = xyz2[g * 3 + 0] - xyz1[point * 3 + 0];
    const float pry = xyz2[g * 3 + 1] - xyz1[point * 3 + 1];
    const float prz = xyz2[g * 3 + 2] - xyz1[point * 3 + 2];
    {
        float* pp = outp + (size_t)(point * Kn + kk) * 3;
        pp[0] = prx; pp[1] = pry; pp[2] = prz;
    }

    // linear_p hidden: relu(bn(pr @ wp1 + bp1))  (3 channels, uniform scalar weights)
    float h[3];
    #pragma unroll
    for (int j = 0; j < 3; ++j) {
        float v = prx * wp1[0 * 3 + j] + pry * wp1[1 * 3 + j] + prz * wp1[2 * 3 + j] + bp1[j];
        const float s = pg[j] * rsqrtf(pv[j] + EPSf);
        v = (v - pm[j]) * s + pb[j];
        h[j] = fmaxf(v, 0.0f);
    }

    // logits: (xk - xq + foldsum(p)) -> bn1 -> relu -> ww1   (64 -> 8)
    float u[8] = {0.f, 0.f, 0.f, 0.f, 0.f, 0.f, 0.f, 0.f};
    const float* xkp = xk + (size_t)g * MIDn;
    #pragma unroll 4
    for (int c = 0; c < MIDn; ++c) {
        const float pf = h[0] * (s_wp2[c]             + s_wp2[c + 64])
                       + h[1] * (s_wp2[OUTn + c]      + s_wp2[OUTn + c + 64])
                       + h[2] * (s_wp2[2 * OUTn + c]  + s_wp2[2 * OUTn + c + 64])
                       + s_bp2[c] + s_bp2[c + 64];
        const float wl = xkp[c] - s_xq[lp][c] + pf;
        const float t  = fmaxf(wl * s_s1[c] + s_t1[c], 0.0f);
        #pragma unroll
        for (int j = 0; j < 8; ++j) u[j] += t * s_ww1[c * 8 + j];
    }

    // bn2 -> relu -> ww2   (8 -> 16)
    float logit[16];
    #pragma unroll
    for (int l = 0; l < 16; ++l) logit[l] = s_bw2[l];
    #pragma unroll
    for (int j = 0; j < 8; ++j) {
        const float uj = fmaxf((u[j] + s_bw1[j]) * s_s2[j] + s_t2[j], 0.0f);
        #pragma unroll
        for (int l = 0; l < 16; ++l) logit[l] += uj * s_ww2[j * 16 + l];
    }

    // softmax over the 16 neighbors (= 16 lanes of this half-wave), per channel
    float wsm[16];
    #pragma unroll
    for (int l = 0; l < 16; ++l) {
        float m = logit[l];
        #pragma unroll
        for (int d = 8; d >= 1; d >>= 1) m = fmaxf(m, __shfl_xor(m, d, 16));
        const float e = __expf(logit[l] - m);
        float s = e;
        #pragma unroll
        for (int d = 8; d >= 1; d >>= 1) s += __shfl_xor(s, d, 16);
        wsm[l] = e / s;
    }

    // grouped weighting: out[c] = (xv[c] + p[c]) * wsm[c % 16]
    const float* xvp = xv + (size_t)g * OUTn;
    float* op = outx + (size_t)(point * Kn + kk) * OUTn;
    #pragma unroll 4
    for (int c = 0; c < OUTn; ++c) {
        const float pc = h[0] * s_wp2[c] + h[1] * s_wp2[OUTn + c]
                       + h[2] * s_wp2[2 * OUTn + c] + s_bp2[c];
        op[c] = (xvp[c] + pc) * wsm[c & 15];
    }
}

// ---------------------------------------------------------------------------
extern "C" void kernel_launch(void* const* d_in, const int* in_sizes, int n_in,
                              void* d_out, int out_size, void* d_ws, size_t ws_size,
                              hipStream_t stream)
{
    const float* xyz1    = (const float*)d_in[0];
    const float* xyz2    = (const float*)d_in[1];
    const float* points1 = (const float*)d_in[2];
    const float* points2 = (const float*)d_in[3];
    const float* wq  = (const float*)d_in[4];  const float* bq  = (const float*)d_in[5];
    const float* wk  = (const float*)d_in[6];  const float* bk  = (const float*)d_in[7];
    const float* wv  = (const float*)d_in[8];  const float* bv  = (const float*)d_in[9];
    const float* wp1 = (const float*)d_in[10]; const float* bp1 = (const float*)d_in[11];
    const float* pg  = (const float*)d_in[12]; const float* pb  = (const float*)d_in[13];
    const float* pm  = (const float*)d_in[14]; const float* pv  = (const float*)d_in[15];
    const float* wp2 = (const float*)d_in[16]; const float* bp2 = (const float*)d_in[17];
    const float* b1g = (const float*)d_in[18]; const float* b1b = (const float*)d_in[19];
    const float* b1m = (const float*)d_in[20]; const float* b1v = (const float*)d_in[21];
    const float* ww1 = (const float*)d_in[22]; const float* bw1 = (const float*)d_in[23];
    const float* b2g = (const float*)d_in[24]; const float* b2b = (const float*)d_in[25];
    const float* b2m = (const float*)d_in[26]; const float* b2v = (const float*)d_in[27];
    const float* ww2 = (const float*)d_in[28]; const float* bw2 = (const float*)d_in[29];

    // workspace layout
    float* xq     = (float*)d_ws;                 // 16384*64
    float* xk     = xq + (size_t)NPTS * MIDn;     // 16384*64
    float* xv     = xk + (size_t)NPTS * MIDn;     // 16384*128
    int*   idxbuf = (int*)(xv + (size_t)NPTS * OUTn);  // 16384*16

    float* outx = (float*)d_out;                       // [n,K,128]
    float* outp = outx + (size_t)NPTS * Kn * OUTn;     // [n,K,3]

    proj_wmma_kernel<<<dim3(NPTS / 16, 16), 32, 0, stream>>>(
        points1, points2, wq, bq, wk, bk, wv, bv, xq, xk, xv);

    knn_kernel<<<dim3(NPTS / TS), TS, 0, stream>>>(xyz1, xyz2, idxbuf);

    attn_kernel<<<dim3(NPTS / 16), 256, 0, stream>>>(
        xyz1, xyz2, xq, xk, xv, idxbuf,
        wp1, bp1, pg, pb, pm, pv, wp2, bp2,
        b1g, b1b, b1m, b1v, ww1, bw1,
        b2g, b2b, b2m, b2v, ww2, bw2,
        outx, outp);
}
